// TransformerLayer_Combined_7679401526017
// MI455X (gfx1250) — matrix operand order repaired and checked by reading, not compiled, not run on hardware
//
#include <hip/hip_runtime.h>
#include <math.h>

typedef __bf16 bf16_t;
typedef __attribute__((ext_vector_type(16))) __bf16 v16bf;
typedef __attribute__((ext_vector_type(8)))  __bf16 v8bf;
typedef __attribute__((ext_vector_type(8)))  float  v8f;

#define S_    1024
#define B_    4
#define H_    1024
#define NH_   16
#define HPH_  64
#define NF_   4
#define NHEADS (B_*NH_)   // 64
#define MROWS  (S_*B_)    // 4096

__device__ inline v16bf cat16(v8bf lo, v8bf hi) {
  return __builtin_shufflevector(lo, hi, 0,1,2,3,4,5,6,7,8,9,10,11,12,13,14,15);
}

// ---------------------------------------------------------------- LayerNorm 1
__global__ __launch_bounds__(256)
void ln1_kernel(const float* __restrict__ x, const float* __restrict__ g,
                const float* __restrict__ b, float* __restrict__ xln,
                bf16_t* __restrict__ xlnb) {
  int row = blockIdx.x;                       // 0..4095 (s*B+b)
  const float* xr = x + (size_t)row * H_;
  float vals[4], s = 0.f, ss = 0.f;
  #pragma unroll
  for (int i = 0; i < 4; ++i) {
    float v = xr[threadIdx.x + i * 256];
    vals[i] = v; s += v; ss += v * v;
  }
  #pragma unroll
  for (int off = 16; off > 0; off >>= 1) {
    s  += __shfl_down(s,  off, 32);
    ss += __shfl_down(ss, off, 32);
  }
  __shared__ float smem[16];
  int w = threadIdx.x >> 5, l = threadIdx.x & 31;
  if (l == 0) { smem[w] = s; smem[8 + w] = ss; }
  __syncthreads();
  if (threadIdx.x == 0) {
    float ts = 0.f, tss = 0.f;
    for (int i = 0; i < 8; ++i) { ts += smem[i]; tss += smem[8 + i]; }
    smem[0] = ts; smem[8] = tss;
  }
  __syncthreads();
  float mean = smem[0] * (1.f / H_);
  float var  = smem[8] * (1.f / H_) - mean * mean;
  float r = rsqrtf(var + 1e-6f);
  #pragma unroll
  for (int i = 0; i < 4; ++i) {
    int c = threadIdx.x + i * 256;
    float o = (vals[i] - mean) * r * g[c] + b[c];
    xln[(size_t)row * H_ + c]  = o;
    xlnb[(size_t)row * H_ + c] = (bf16_t)o;
  }
}

// ------------------------------------------------- pack B operand fragments
// dst layout: [batch][kt][nt][lane(32)][16] bf16; lane<16 -> K 0..15 of col n,
// lane>=16 -> K 16..31 of col n  (contiguous k order, same as the A fragments).
template <typename T>
__global__ void pack_b_kernel(const T* __restrict__ src, long batchStride,
                              int rowStride, bf16_t* __restrict__ dst,
                              int K, int N, int nBatch) {
  int Kt = K >> 5, Nt = N >> 4;
  long tid = (long)blockIdx.x * blockDim.x + threadIdx.x;
  long total = (long)nBatch * Kt * Nt * 32;
  if (tid >= total) return;
  int lane = (int)(tid & 31);
  long t = tid >> 5;
  int nt = (int)(t % Nt); t /= Nt;
  int kt = (int)(t % Kt); int batch = (int)(t / Kt);
  const T* s = src + (long)batch * batchStride;
  int n = nt * 16 + (lane & 15);
  int kbase = kt * 32 + ((lane >> 4) << 4);
  bf16_t* d = dst + ((((long)batch * Kt + kt) * Nt + nt) * 32 + lane) * 16;
  #pragma unroll
  for (int j = 0; j < 16; ++j)
    d[j] = (bf16_t)(float)s[(long)(kbase + j) * rowStride + n];
}

// ----------------------------------------------------------- WMMA bf16 GEMM
// C[M,N] = A[M,K](bf16,row-major,lda) @ packedB + bias.  Wave = 32x64 tile.
template <bool OUT_BF16>
__global__ __launch_bounds__(256)
void gemm_bf16_kernel(const bf16_t* __restrict__ A, int lda,
                      const bf16_t* __restrict__ Bp,
                      const float* __restrict__ bias,
                      void* __restrict__ C, int ldc,
                      int M, int N, int K) {
  int lane = threadIdx.x & 31;
  int w = threadIdx.x >> 5;
  int wm = w & 1, wn = w >> 1;
  int tileM = blockIdx.y * 64 + wm * 32;
  int tileN = blockIdx.x * 256 + wn * 64;
  int Nt = N >> 4;
  int halfsel = lane >> 4;
  int rowA0 = tileM + (lane & 15);
  int aoff = halfsel << 4;
  v8f acc[2][4] = {};
  for (int kk = 0; kk < K; kk += 32) {
    int kt = kk >> 5;
    v16bf a[2];
    #pragma unroll
    for (int i = 0; i < 2; ++i) {
      const bf16_t* pa = A + (long)(rowA0 + i * 16) * lda + kk + aoff;
      a[i] = cat16(*(const v8bf*)pa, *(const v8bf*)(pa + 8));
    }
    const bf16_t* pb0 = Bp + (((long)kt * Nt + (tileN >> 4)) * 32 + lane) * 16;
    #pragma unroll
    for (int j = 0; j < 4; ++j) {
      v16bf bfr = *(const v16bf*)(pb0 + (long)j * 512);
      acc[0][j] = __builtin_amdgcn_wmma_f32_16x16x32_bf16(
          false, a[0], false, bfr, (short)0, acc[0][j], false, false);
      acc[1][j] = __builtin_amdgcn_wmma_f32_16x16x32_bf16(
          false, a[1], false, bfr, (short)0, acc[1][j], false, false);
    }
  }
  #pragma unroll
  for (int i = 0; i < 2; ++i)
    #pragma unroll
    for (int j = 0; j < 4; ++j) {
      int colg = tileN + j * 16 + (lane & 15);
      float bv = bias[colg];
      #pragma unroll
      for (int r = 0; r < 8; ++r) {
        int rowg = tileM + i * 16 + r + (halfsel << 3);
        float v = acc[i][j][r] + bv;
        if (OUT_BF16) ((bf16_t*)C)[(long)rowg * ldc + colg] = (bf16_t)v;
        else          ((float*)C)[(long)rowg * ldc + colg] = v;
      }
    }
}

// ------------------------------------------------- row sums of squares (q2/k2)
__global__ void rowsq_kernel(const bf16_t* __restrict__ src, float* __restrict__ dst) {
  int t = blockIdx.x * blockDim.x + threadIdx.x;     // (h,s): 65536
  if (t >= NHEADS * S_) return;
  int s = t & (S_ - 1);
  int h = t >> 10;
  int b = h >> 4, nh = h & 15;
  const bf16_t* p = src + (long)(s * B_ + b) * H_ + nh * HPH_;
  float acc = 0.f;
  #pragma unroll 8
  for (int d = 0; d < HPH_; ++d) { float v = (float)p[d]; acc += v * v; }
  dst[t] = acc;                                      // dst[h*S + s]
}

// --------------------------------------------------------- fused attention
// scores = -(q2+k2-2*q.kT)*0.5*NORM_FACTOR ; mres store ; probs=exp ; ctx=probs@V
__global__ __launch_bounds__(256)
void attn_kernel(const bf16_t* __restrict__ qb, const bf16_t* __restrict__ kb,
                 const bf16_t* __restrict__ vp,
                 const float* __restrict__ q2, const float* __restrict__ k2,
                 float* __restrict__ mres, float* __restrict__ ctx) {
  __shared__ __align__(16) bf16_t lds[8][16 * 40 + 8];
  int lane = threadIdx.x & 31;
  int w = threadIdx.x >> 5;
  int h = blockIdx.y;
  int b = h >> 4, nh = h & 15;
  int r0 = blockIdx.x * 128 + w * 16;                 // s-row base in head
  int halfsel = lane >> 4;
  const float kscale = -0.5f * 0.125f;                // -0.5 / sqrt(64)

  // Q as A-operand (16 rows x 64 K): two fragments
  v16bf qa[2];
  {
    const bf16_t* pq = qb + (long)((r0 + (lane & 15)) * B_ + b) * H_
                          + nh * HPH_ + (halfsel << 4);
    #pragma unroll
    for (int i = 0; i < 2; ++i)
      qa[i] = cat16(*(const v8bf*)(pq + i * 32), *(const v8bf*)(pq + i * 32 + 8));
  }
  float q2v[8];
  #pragma unroll
  for (int r = 0; r < 8; ++r) q2v[r] = q2[h * S_ + r0 + r + (halfsel << 3)];

  v8f ctxacc[4] = {};
  bf16_t* myl = &lds[w][0];

  for (int t0 = 0; t0 < S_; t0 += 32) {
    #pragma unroll
    for (int half = 0; half < 2; ++half) {
      int tt = t0 + half * 16;
      int trow = tt + (lane & 15);
      const bf16_t* pk = kb + (long)(trow * B_ + b) * H_ + nh * HPH_ + (halfsel << 4);
      v16bf kf0 = *(const v16bf*)pk;
      v16bf kf1 = *(const v16bf*)(pk + 32);
      v8f c = {};
      c = __builtin_amdgcn_wmma_f32_16x16x32_bf16(false, qa[0], false, kf0,
                                                  (short)0, c, false, false);
      c = __builtin_amdgcn_wmma_f32_16x16x32_bf16(false, qa[1], false, kf1,
                                                  (short)0, c, false, false);
      float k2v = k2[h * S_ + trow];
      float* pm = mres + ((long)h * S_ + r0 + (halfsel << 3)) * S_ + trow;
      #pragma unroll
      for (int r = 0; r < 8; ++r) {
        float sc = (q2v[r] + k2v - 2.f * c[r]) * kscale;
        pm[(long)r * S_] = sc;
        myl[(r + (halfsel << 3)) * 40 + half * 16 + (lane & 15)] = (bf16_t)__expf(sc);
      }
    }
    // probs tile (16x32) from LDS as A-operand, multiply by packed V
    const bf16_t* pp = myl + (lane & 15) * 40 + (halfsel << 4);
    v16bf pa = cat16(*(const v8bf*)pp, *(const v8bf*)(pp + 8));
    int kt = t0 >> 5;
    const bf16_t* pv = vp + (((long)h * 32 + kt) * 4) * 512 + lane * 16;
    #pragma unroll
    for (int nt = 0; nt < 4; ++nt) {
      v16bf vb = *(const v16bf*)(pv + (long)nt * 512);
      ctxacc[nt] = __builtin_amdgcn_wmma_f32_16x16x32_bf16(
          false, pa, false, vb, (short)0, ctxacc[nt], false, false);
    }
  }
  #pragma unroll
  for (int nt = 0; nt < 4; ++nt)
    #pragma unroll
    for (int r = 0; r < 8; ++r) {
      int srow = r0 + r + (halfsel << 3);
      ctx[(long)(srow * B_ + b) * H_ + nh * HPH_ + nt * 16 + (lane & 15)] =
          ctxacc[nt][r];
    }
}

// --------------------------------------------- residual + per-head LayerNorm2
__global__ void ln2_kernel(const float* __restrict__ xln, const float* __restrict__ ctx,
                           const float* __restrict__ g2, const float* __restrict__ b2,
                           bf16_t* __restrict__ outb) {
  int t = blockIdx.x * blockDim.x + threadIdx.x;     // (sb, nh): 65536
  if (t >= MROWS * NH_) return;
  int nh = t & 15;
  int sb = t >> 4;
  const float* px = xln + (long)sb * H_ + nh * HPH_;
  const float* pc = ctx + (long)sb * H_ + nh * HPH_;
  float v[HPH_], s = 0.f;
  #pragma unroll 8
  for (int d = 0; d < HPH_; ++d) { v[d] = px[d] + pc[d]; s += v[d]; }
  float m = s * (1.f / HPH_), var = 0.f;
  #pragma unroll 8
  for (int d = 0; d < HPH_; ++d) { float u = v[d] - m; var += u * u; }
  var *= (1.f / HPH_);
  float r = rsqrtf(var + 1e-6f);
  bf16_t* po = outb + (long)sb * H_ + nh * HPH_;
  #pragma unroll 8
  for (int d = 0; d < HPH_; ++d) po[d] = (bf16_t)((v[d] - m) * r * g2[d] + b2[d]);
}

// ----------------------------------------------------------------- GLU gate
__global__ void glu_kernel(const bf16_t* __restrict__ h, bf16_t* __restrict__ o) {
  long t = (long)blockIdx.x * blockDim.x + threadIdx.x;  // sb*4*1024
  if (t >= (long)MROWS * NF_ * 1024) return;
  int i = (int)(t & 1023);
  long gg = t >> 10;
  int g = (int)(gg & 3);
  long sb = gg >> 2;
  const bf16_t* ph = h + sb * 8192 + g * 2048;
  float xh = (float)ph[i], yh = (float)ph[1024 + i];
  o[t] = (bf16_t)(xh * (yh > 0.f ? yh : 0.f));
}

// =============================================================== launcher
extern "C" void kernel_launch(void* const* d_in, const int* in_sizes, int n_in,
                              void* d_out, int out_size, void* d_ws, size_t ws_size,
                              hipStream_t stream) {
  (void)in_sizes; (void)n_in; (void)out_size; (void)ws_size;
  const float* hs  = (const float*)d_in[0];
  const float* g1  = (const float*)d_in[1];
  const float* b1  = (const float*)d_in[2];
  const float* g2  = (const float*)d_in[3];
  const float* b2  = (const float*)d_in[4];
  const float* Wq  = (const float*)d_in[5];
  const float* bq  = (const float*)d_in[6];
  const float* Wk  = (const float*)d_in[7];
  const float* bk  = (const float*)d_in[8];
  const float* Wv  = (const float*)d_in[9];
  const float* bv  = (const float*)d_in[10];
  const float* Win = (const float*)d_in[11];
  const float* bin = (const float*)d_in[12];
  const float* Wout= (const float*)d_in[13];
  const float* bout= (const float*)d_in[14];
  const float* Wd  = (const float*)d_in[15];
  const float* bd  = (const float*)d_in[16];

  float* out_f = (float*)d_out;                       // [S,B,H]
  float* mres  = out_f + (size_t)MROWS * H_;          // [64,1024,1024]

  char* ws = (char*)d_ws;
  auto alloc = [&](size_t bytes) -> char* {
    char* p = ws; ws += (bytes + 255) & ~(size_t)255; return p;
  };
  float*  xln   = (float*)alloc((size_t)MROWS * H_ * 4);
  bf16_t* xlnb  = (bf16_t*)alloc((size_t)MROWS * H_ * 2);
  bf16_t* qbuf  = (bf16_t*)alloc((size_t)MROWS * H_ * 2);
  bf16_t* kbuf  = (bf16_t*)alloc((size_t)MROWS * H_ * 2);
  bf16_t* vbuf  = (bf16_t*)alloc((size_t)MROWS * H_ * 2);
  bf16_t* vpack = (bf16_t*)alloc((size_t)NHEADS * 32 * 4 * 512 * 2);
  float*  q2    = (float*)alloc((size_t)NHEADS * S_ * 4);
  float*  k2    = (float*)alloc((size_t)NHEADS * S_ * 4);
  float*  ctx   = (float*)alloc((size_t)MROWS * H_ * 4);
  bf16_t* ctxln = (bf16_t*)alloc((size_t)MROWS * H_ * 2);
  bf16_t* hbuf  = (bf16_t*)alloc((size_t)MROWS * 8192 * 2);
  bf16_t* glu   = (bf16_t*)alloc((size_t)MROWS * 4096 * 2);
  bf16_t* og    = (bf16_t*)alloc((size_t)MROWS * H_ * 2);
  bf16_t* wqp   = (bf16_t*)alloc((size_t)H_ * H_ * 2);
  bf16_t* wkp   = (bf16_t*)alloc((size_t)H_ * H_ * 2);
  bf16_t* wvp   = (bf16_t*)alloc((size_t)H_ * H_ * 2);
  bf16_t* wdp   = (bf16_t*)alloc((size_t)H_ * H_ * 2);
  bf16_t* winp  = (bf16_t*)alloc((size_t)NF_ * 256 * 2048 * 2);
  bf16_t* woutp = (bf16_t*)alloc((size_t)NF_ * 1024 * 256 * 2);

  // 1) LN1 (+ bf16 copy)
  ln1_kernel<<<MROWS, 256, 0, stream>>>(hs, g1, b1, xln, xlnb);

  // 2) pack weights
  pack_b_kernel<float><<<256, 256, 0, stream>>>(Wq, 0L, H_, wqp, H_, H_, 1);
  pack_b_kernel<float><<<256, 256, 0, stream>>>(Wk, 0L, H_, wkp, H_, H_, 1);
  pack_b_kernel<float><<<256, 256, 0, stream>>>(Wv, 0L, H_, wvp, H_, H_, 1);
  pack_b_kernel<float><<<256, 256, 0, stream>>>(Wd, 0L, H_, wdp, H_, H_, 1);
  pack_b_kernel<float><<<512, 256, 0, stream>>>(Win, (long)256 * 2048, 2048,
                                                winp, 256, 2048, NF_);
  pack_b_kernel<float><<<256, 256, 0, stream>>>(Wout, (long)1024 * 256, 256,
                                                woutp, 1024, 256, NF_);

  // 3) QKV projections (bf16 out)
  dim3 gqkv(H_ / 256, MROWS / 64);
  gemm_bf16_kernel<true><<<gqkv, 256, 0, stream>>>(xlnb, H_, wqp, bq, qbuf, H_,
                                                   MROWS, H_, H_);
  gemm_bf16_kernel<true><<<gqkv, 256, 0, stream>>>(xlnb, H_, wkp, bk, kbuf, H_,
                                                   MROWS, H_, H_);
  gemm_bf16_kernel<true><<<gqkv, 256, 0, stream>>>(xlnb, H_, wvp, bv, vbuf, H_,
                                                   MROWS, H_, H_);

  // 4) q2 / k2
  rowsq_kernel<<<256, 256, 0, stream>>>(qbuf, q2);
  rowsq_kernel<<<256, 256, 0, stream>>>(kbuf, k2);

  // 5) pack V per head (head h offset = h*64, row stride B*H)
  pack_b_kernel<bf16_t><<<1024, 256, 0, stream>>>(vbuf, 64L, B_ * H_, vpack,
                                                  S_, HPH_, NHEADS);

  // 6) fused attention: scores -> mres, exp -> probs@V -> ctx
  attn_kernel<<<dim3(8, NHEADS), 256, 0, stream>>>(qbuf, kbuf, vpack, q2, k2,
                                                   mres, ctx);

  // 7) residual + per-head LN2 -> bf16
  ln2_kernel<<<256, 256, 0, stream>>>(xln, ctx, g2, b2, ctxln);

  // 8) grouped FFN dense_in (bf16 out) then GLU
  for (int g = 0; g < NF_; ++g) {
    gemm_bf16_kernel<true><<<dim3(2048 / 256, MROWS / 64), 256, 0, stream>>>(
        ctxln + g * 256, H_, winp + (size_t)g * 8 * 128 * 512, bin + g * 2048,
        hbuf + g * 2048, 8192, MROWS, 2048, 256);
  }
  glu_kernel<<<65536, 256, 0, stream>>>(hbuf, glu);

  // 9) grouped FFN dense_out (bf16 out)
  for (int g = 0; g < NF_; ++g) {
    gemm_bf16_kernel<true><<<dim3(1, MROWS / 64), 256, 0, stream>>>(
        glu + g * 1024, 4096, woutp + (size_t)g * 32 * 16 * 512, bout + g * 256,
        og + g * 256, H_, MROWS, 256, 1024);
  }

  // 10) output dense (fp32 out -> d_out)
  gemm_bf16_kernel<false><<<dim3(H_ / 256, MROWS / 64), 256, 0, stream>>>(
      og, H_, wdp, bd, out_f, H_, MROWS, H_, H_);
}
